// MultiHeadAttention_45672682226566
// MI455X (gfx1250) — compile-verified, matrix operand order checked by reading, not run on hardware
//
#include <hip/hip_runtime.h>
#include <hip/hip_bf16.h>

// ---------------------------------------------------------------------------
// Types for CDNA5 WMMA (wave32, 16x16x32 bf16 -> f32)
// ---------------------------------------------------------------------------
typedef __attribute__((ext_vector_type(16))) __bf16 v16bf;
typedef __attribute__((ext_vector_type(8)))  float  v8f;

union FragAB { uint4 u[2]; v16bf v; };   // 32 bytes: one A or B fragment
union FragC  { v8f v; float f[8]; };     // 16x16 f32 C/D accumulator

__device__ __forceinline__ unsigned short f32_to_bf16(float f) {
  unsigned int u = __float_as_uint(f);
  u += 0x7fffu + ((u >> 16) & 1u);       // round-to-nearest-even
  return (unsigned short)(u >> 16);
}
__device__ __forceinline__ float bf16_to_f32(unsigned short h) {
  return __uint_as_float(((unsigned int)h) << 16);
}

// ---------------------------------------------------------------------------
// CDNA5 async global->LDS copy (ASYNCcnt-tracked, bypasses VGPRs)
//   LDS offset = low 32 bits of the flat shared-memory address (ISA 10.2)
// ---------------------------------------------------------------------------
__device__ __forceinline__ void async_ld16(const unsigned short* lds,
                                           const unsigned short* g) {
  unsigned l32 = (unsigned)(unsigned long long)lds;
  unsigned long long g64 = (unsigned long long)g;
  asm volatile("global_load_async_to_lds_b128 %0, %1, off"
               :: "v"(l32), "v"(g64) : "memory");
}
__device__ __forceinline__ void wait_async0() {
  asm volatile("s_wait_asynccnt 0" ::: "memory");
}

// ---------------------------------------------------------------------------
// Problem constants
// ---------------------------------------------------------------------------
#define BB   4
#define SS   1024
#define HID  1024
#define HH   16
#define DD   64
#define NEG_BIG (-1.0e30f)

// ---------------------------------------------------------------------------
// Kernel 1a: f32 -> bf16 conversion (x)
// ---------------------------------------------------------------------------
__global__ __launch_bounds__(256)
void cvt_bf16_kernel(const float* __restrict__ in, unsigned short* __restrict__ out, int n) {
  int i = blockIdx.x * blockDim.x + threadIdx.x;
  if (i < n) out[i] = f32_to_bf16(in[i]);
}

// ---------------------------------------------------------------------------
// Kernel 1b: f32 [K][N] -> bf16 [N][K] (weights, K = 1024)
//   K-major output lets GEMM B-tiles stream as contiguous async copies
// ---------------------------------------------------------------------------
__global__ __launch_bounds__(256)
void cvt_transpose_kernel(const float* __restrict__ in, unsigned short* __restrict__ out,
                          int N) {
  int idx = blockIdx.x * blockDim.x + threadIdx.x;   // k fastest
  int k = idx & 1023;
  int n = idx >> 10;
  if (n < N) out[(size_t)n * 1024 + k] = f32_to_bf16(in[(size_t)k * N + n]);
}

// ---------------------------------------------------------------------------
// Kernel 2: bf16 WMMA GEMM  C[M,N] = A[M,K] * Bt[N,K]^T
//   block tile 128x128, 8 waves (4x2), wave tile 32x64, K-step 32
//   double-buffered LDS, async global->LDS staging
// ---------------------------------------------------------------------------
#define GTM 128
#define GTN 128
#define GTK 32

__global__ __launch_bounds__(256)
void gemm_bf16_kernel(const unsigned short* __restrict__ A,    // [M][K]
                      const unsigned short* __restrict__ Bt,   // [N][K]
                      float* __restrict__ Cf,
                      unsigned short* __restrict__ Cb,
                      int M, int N, int K) {
  __shared__ unsigned short lA[2][GTM * GTK];   // 2 x 8 KB  [row][k]
  __shared__ unsigned short lB[2][GTN * GTK];   // 2 x 8 KB  [n][k]
  const int bm = blockIdx.x, bn = blockIdx.y;
  const int t = threadIdx.x;
  const int lane = t & 31, wave = t >> 5;
  const int wm = wave & 3, wn = wave >> 2;      // 4x2 wave grid
  const int lhalf = lane >> 4, lmod = lane & 15;
  const int kb8 = lhalf * 8;

  // per-thread staging: each thread copies 2 chunks of A and 2 of B per fill
  const int cr = t >> 2;                         // 0..63
  const int cc = (t & 3) * 8;                    // 0,8,16,24
  const unsigned short* aG0 = A  + (size_t)(bm * GTM + cr) * K + cc;
  const unsigned short* aG1 = aG0 + (size_t)64 * K;
  const unsigned short* bG0 = Bt + (size_t)(bn * GTN + cr) * K + cc;
  const unsigned short* bG1 = bG0 + (size_t)64 * K;
  const int lOff = cr * GTK + cc;

  FragC acc[2][4];
#pragma unroll
  for (int ma = 0; ma < 2; ++ma)
#pragma unroll
    for (int nt = 0; nt < 4; ++nt)
#pragma unroll
      for (int j = 0; j < 8; ++j) acc[ma][nt].f[j] = 0.0f;

  const int niter = K / GTK;
  // prologue: fill buffer 0
  async_ld16(&lA[0][lOff], aG0);
  async_ld16(&lA[0][lOff + 64 * GTK], aG1);
  async_ld16(&lB[0][lOff], bG0);
  async_ld16(&lB[0][lOff + 64 * GTK], bG1);

  for (int it = 0; it < niter; ++it) {
    wait_async0();          // my fills of buf (it&1) complete
    __syncthreads();        // everyone's fills done; prev compute done
    if (it + 1 < niter) {
      const size_t ko = (size_t)(it + 1) * GTK;
      const int nb = (it + 1) & 1;
      async_ld16(&lA[nb][lOff], aG0 + ko);
      async_ld16(&lA[nb][lOff + 64 * GTK], aG1 + ko);
      async_ld16(&lB[nb][lOff], bG0 + ko);
      async_ld16(&lB[nb][lOff + 64 * GTK], bG1 + ko);
    }
    const unsigned short* sA = lA[it & 1];
    const unsigned short* sB = lB[it & 1];

    FragAB af[2];
#pragma unroll
    for (int ma = 0; ma < 2; ++ma) {
      const unsigned short* ap = sA + (wm * 32 + ma * 16 + lmod) * GTK + kb8;
      af[ma].u[0] = *(const uint4*)(ap);
      af[ma].u[1] = *(const uint4*)(ap + 16);
    }
#pragma unroll
    for (int nt = 0; nt < 4; ++nt) {
      FragAB bf;
      const unsigned short* bp = sB + (wn * 64 + nt * 16 + lmod) * GTK + kb8;
      bf.u[0] = *(const uint4*)(bp);
      bf.u[1] = *(const uint4*)(bp + 16);
#pragma unroll
      for (int ma = 0; ma < 2; ++ma)
        acc[ma][nt].v = __builtin_amdgcn_wmma_f32_16x16x32_bf16(
            false, af[ma].v, false, bf.v, (short)0, acc[ma][nt].v, false, false);
    }
  }

  // epilogue: element (row = j + 8*lhalf, col = lmod) per C-layout
#pragma unroll
  for (int ma = 0; ma < 2; ++ma) {
#pragma unroll
    for (int nt = 0; nt < 4; ++nt) {
#pragma unroll
      for (int j = 0; j < 8; ++j) {
        int row = bm * GTM + wm * 32 + ma * 16 + j + 8 * lhalf;
        int col = bn * GTN + wn * 64 + nt * 16 + lmod;
        if (Cf) Cf[(size_t)row * N + col] = acc[ma][nt].f[j];
        else    Cb[(size_t)row * N + col] = f32_to_bf16(acc[ma][nt].f[j]);
      }
    }
  }
}

// ---------------------------------------------------------------------------
// Kernel 3: RoPE (split-half) in-place on q,k + q scaling
// ---------------------------------------------------------------------------
__global__ __launch_bounds__(256)
void rope_kernel(unsigned short* __restrict__ qkv) {
  int idx = blockIdx.x * blockDim.x + threadIdx.x;  // 4*1024*2*16*32 threads
  int d  = idx & 31;
  int h  = (idx >> 5) & 15;
  int w  = (idx >> 9) & 1;          // 0 = q, 1 = k
  int s  = (idx >> 10) & 1023;
  int b  = idx >> 20;
  size_t base = (((size_t)b * SS + s) * 3 + w) * (HH * DD) + (size_t)h * DD;
  float first  = bf16_to_f32(qkv[base + d]);
  float second = bf16_to_f32(qkv[base + d + 32]);
  float inv_ts = __powf(10000.0f, -(float)d * (1.0f / 32.0f));
  float sinus  = (float)s * inv_ts;
  float sn, cs;
  __sincosf(sinus, &sn, &cs);
  float o1 = first * cs - second * sn;
  float o2 = second * cs + first * sn;
  if (w == 0) { o1 *= 0.125f; o2 *= 0.125f; }   // 1/sqrt(64)
  qkv[base + d]      = f32_to_bf16(o1);
  qkv[base + d + 32] = f32_to_bf16(o2);
}

// ---------------------------------------------------------------------------
// Kernel 3b: V transpose  qkv[B,S,3,H,D] -> vT[B,H,D,S]
//   d-major V lets attention V-tiles stream as contiguous async copies
// ---------------------------------------------------------------------------
__global__ __launch_bounds__(256)
void transpose_v_kernel(const unsigned short* __restrict__ qkv,
                        unsigned short* __restrict__ vT) {
  int idx = blockIdx.x * blockDim.x + threadIdx.x;  // B*H*D*S = 4M threads
  int s = idx & 1023;
  int d = (idx >> 10) & 63;
  int h = (idx >> 16) & 15;
  int b = idx >> 20;
  size_t src = (((size_t)b * SS + s) * 3 + 2) * (HH * DD) + (size_t)h * DD + d;
  vT[(((size_t)b * HH + h) * DD + d) * SS + s] = qkv[src];
}

// ---------------------------------------------------------------------------
// Kernel 4: causal flash attention, WMMA for QK^T and PV
//   grid (S/64, H, B), 128 threads = 4 waves, wave owns 16x64 Q tile
//   double-buffered async K/V staging
// ---------------------------------------------------------------------------
__global__ __launch_bounds__(128)
void attn_kernel(const unsigned short* __restrict__ qkv,
                 const unsigned short* __restrict__ vT,
                 unsigned short* __restrict__ ctx) {
  __shared__ unsigned short lK[2][32 * 64];     // [key][d]   2 x 4 KB
  __shared__ unsigned short lV[2][64 * 32];     // [d][key]   2 x 4 KB
  __shared__ unsigned short lP[4][16 * 32];     // per-wave P 4 KB

  const int b = blockIdx.z, h = blockIdx.y, qblk = blockIdx.x;
  const int t = threadIdx.x;
  const int lane = t & 31, wave = t >> 5;
  const int lhalf = lane >> 4, lmod = lane & 15;
  const int kb8 = lhalf * 8;
  const int q0 = qblk * 64 + wave * 16;
  const size_t keyStride = 3 * HH * DD;         // qkv stride per sequence pos

  auto qkvIdx = [&](int s, int which, int d) -> size_t {
    return (((size_t)b * SS + s) * 3 + which) * (HH * DD) + (size_t)h * DD + d;
  };

  // Q fragments (2 x 16x32 over D=64) held in registers
  FragAB qf[2];
  {
    int qr = q0 + lmod;
#pragma unroll
    for (int f = 0; f < 2; ++f) {
      qf[f].u[0] = *(const uint4*)(qkv + qkvIdx(qr, 0, f * 32 + kb8));
      qf[f].u[1] = *(const uint4*)(qkv + qkvIdx(qr, 0, f * 32 + kb8 + 16));
    }
  }

  // per-thread async staging addresses (2 chunks K, 2 chunks V per fill)
  const unsigned short* kG0 = qkv + qkvIdx(t >> 3, 1, (t & 7) * 8);
  const unsigned short* kG1 = kG0 + 16 * keyStride;
  const int lKoff = (t >> 3) * 64 + (t & 7) * 8;
  const unsigned short* vG0 = vT + (((size_t)b * HH + h) * DD + (t >> 2)) * SS + (t & 3) * 8;
  const unsigned short* vG1 = vG0 + (size_t)32 * SS;
  const int lVoff = (t >> 2) * 32 + (t & 3) * 8;

  FragC accO[4];
  float m[8], l[8];
#pragma unroll
  for (int nt = 0; nt < 4; ++nt)
#pragma unroll
    for (int j = 0; j < 8; ++j) accO[nt].f[j] = 0.0f;
#pragma unroll
  for (int j = 0; j < 8; ++j) { m[j] = NEG_BIG; l[j] = 0.0f; }

  const int niter = (qblk * 64 + 64) / 32;      // causal bound, >= 2
  auto issueKV = [&](int it, int buf) {
    size_t kOff = (size_t)it * 32 * keyStride;  // 32 keys forward in qkv
    size_t vOff = (size_t)it * 32;              // 32 keys forward in vT
    async_ld16(&lK[buf][lKoff],            kG0 + kOff);
    async_ld16(&lK[buf][lKoff + 16 * 64],  kG1 + kOff);
    async_ld16(&lV[buf][lVoff],            vG0 + vOff);
    async_ld16(&lV[buf][lVoff + 32 * 32],  vG1 + vOff);
  };
  issueKV(0, 0);

  for (int it = 0; it < niter; ++it) {
    const int kb = it * 32;
    wait_async0();
    __syncthreads();
    if (it + 1 < niter) issueKV(it + 1, (it + 1) & 1);
    const unsigned short* sK = lK[it & 1];
    const unsigned short* sV = lV[it & 1];

    // S = Q K^T : two 16-key column tiles, chained over D
    FragC st[2];
#pragma unroll
    for (int nt = 0; nt < 2; ++nt) {
#pragma unroll
      for (int j = 0; j < 8; ++j) st[nt].f[j] = 0.0f;
#pragma unroll
      for (int f = 0; f < 2; ++f) {
        FragAB bf;   // lane holds key col (nt*16+lmod), contiguous d chunks
        const unsigned short* kp = sK + (nt * 16 + lmod) * 64 + f * 32 + kb8;
        bf.u[0] = *(const uint4*)(kp);
        bf.u[1] = *(const uint4*)(kp + 16);
        st[nt].v = __builtin_amdgcn_wmma_f32_16x16x32_bf16(
            false, qf[f].v, false, bf.v, (short)0, st[nt].v, false, false);
      }
    }

    // causal mask + online softmax (C-layout row = j + 8*lhalf)
#pragma unroll
    for (int j = 0; j < 8; ++j) {
      int qi = q0 + j + 8 * lhalf;
#pragma unroll
      for (int nt = 0; nt < 2; ++nt) {
        int ki = kb + nt * 16 + lmod;
        if (ki > qi) st[nt].f[j] = NEG_BIG;
      }
      float mx = fmaxf(st[0].f[j], st[1].f[j]);
#pragma unroll
      for (int off = 8; off >= 1; off >>= 1)
        mx = fmaxf(mx, __shfl_xor(mx, off, 32));       // 16-lane row reduce
      float mnew  = fmaxf(m[j], mx);
      float alpha = __expf(m[j] - mnew);
      m[j] = mnew;
      float p0 = __expf(st[0].f[j] - mnew);
      float p1 = __expf(st[1].f[j] - mnew);
      st[0].f[j] = p0; st[1].f[j] = p1;
      float rs = p0 + p1;
#pragma unroll
      for (int off = 8; off >= 1; off >>= 1)
        rs += __shfl_xor(rs, off, 32);
      l[j] = l[j] * alpha + rs;
#pragma unroll
      for (int nt = 0; nt < 4; ++nt) accO[nt].f[j] *= alpha;
    }

    // P: C layout -> A layout via per-wave LDS round trip
    unsigned short* P = lP[wave];
#pragma unroll
    for (int nt = 0; nt < 2; ++nt)
#pragma unroll
      for (int j = 0; j < 8; ++j)
        P[(j + 8 * lhalf) * 32 + nt * 16 + lmod] = f32_to_bf16(st[nt].f[j]);
    asm volatile("s_wait_dscnt 0" ::: "memory");   // DS in-order per wave
    FragAB pf;
    pf.u[0] = *(const uint4*)(P + lmod * 32 + kb8);
    pf.u[1] = *(const uint4*)(P + lmod * 32 + kb8 + 16);

    // O += P V : 4 d-tiles of 16
#pragma unroll
    for (int nt = 0; nt < 4; ++nt) {
      FragAB vf;
      const unsigned short* vp = sV + (nt * 16 + lmod) * 32 + kb8;
      vf.u[0] = *(const uint4*)(vp);
      vf.u[1] = *(const uint4*)(vp + 16);
      accO[nt].v = __builtin_amdgcn_wmma_f32_16x16x32_bf16(
          false, pf.v, false, vf.v, (short)0, accO[nt].v, false, false);
    }
  }

  // normalize and store ctx [B][S][H*D] bf16
#pragma unroll
  for (int nt = 0; nt < 4; ++nt) {
#pragma unroll
    for (int j = 0; j < 8; ++j) {
      int qi = q0 + j + 8 * lhalf;
      int d  = nt * 16 + lmod;
      float v = accO[nt].f[j] / l[j];
      ctx[((size_t)b * SS + qi) * (HH * DD) + (size_t)h * DD + d] = f32_to_bf16(v);
    }
  }
}

// ---------------------------------------------------------------------------
// Host-side launcher
// ---------------------------------------------------------------------------
extern "C" void kernel_launch(void* const* d_in, const int* in_sizes, int n_in,
                              void* d_out, int out_size, void* d_ws, size_t ws_size,
                              hipStream_t stream) {
  const float* x    = (const float*)d_in[0];   // [4,1024,1024]
  const float* wqkv = (const float*)d_in[1];   // [1024,3072]
  const float* wo   = (const float*)d_in[2];   // [1024,1024]
  float* out = (float*)d_out;                  // [4,1024,1024] f32

  // workspace layout (bf16 shorts): 56 MB total
  unsigned short* xb     = (unsigned short*)d_ws;                 // 4M  [4096,1024]
  unsigned short* wqkvT  = xb    + (size_t)BB * SS * HID;         // 3M  [3072,1024]
  unsigned short* woT    = wqkvT + (size_t)3 * HH * DD * HID;     // 1M  [1024,1024]
  unsigned short* qkv    = woT   + (size_t)HID * HH * DD;         // 12M [B,S,3,H,D]
  unsigned short* ctx    = qkv   + (size_t)BB * SS * 3 * HH * DD; // 4M  [B,S,H*D]
  unsigned short* vT     = ctx   + (size_t)BB * SS * HH * DD;     // 4M  [B,H,D,S]

  const int nx  = BB * SS * HID;         // 4194304
  const int nw1 = HID * 3 * HH * DD;     // 3145728
  const int nw2 = HID * HH * DD;         // 1048576
  cvt_bf16_kernel<<<(nx + 255) / 256, 256, 0, stream>>>(x, xb, nx);
  cvt_transpose_kernel<<<(nw1 + 255) / 256, 256, 0, stream>>>(wqkv, wqkvT, 3 * HH * DD);
  cvt_transpose_kernel<<<(nw2 + 255) / 256, 256, 0, stream>>>(wo, woT, HID);

  // QKV projection: [4096,1024] x [1024,3072] -> bf16 qkv
  dim3 g1((BB * SS) / GTM, (3 * HH * DD) / GTN);   // (32, 24)
  gemm_bf16_kernel<<<g1, 256, 0, stream>>>(xb, wqkvT, nullptr, qkv,
                                           BB * SS, 3 * HH * DD, HID);

  // RoPE on q,k (+ q scaling), in place; then V transpose
  rope_kernel<<<(BB * SS * 2 * HH * 32) / 256, 256, 0, stream>>>(qkv);
  transpose_v_kernel<<<(BB * SS * HH * DD) / 256, 256, 0, stream>>>(qkv, vT);

  // flash attention
  dim3 ga(SS / 64, HH, BB);                        // (16, 16, 4)
  attn_kernel<<<ga, 128, 0, stream>>>(qkv, vT, ctx);

  // output projection: [4096,1024] x [1024,1024] -> f32 out
  dim3 g2((BB * SS) / GTM, HID / GTN);             // (32, 8)
  gemm_bf16_kernel<<<g2, 256, 0, stream>>>(ctx, woT, out, nullptr,
                                           BB * SS, HID, HH * DD);
}
